// Net_13726715478792
// MI455X (gfx1250) — compile-verified, matrix operand order checked by reading, not run on hardware
//
#include <hip/hip_runtime.h>
#include <math.h>

// ---------------- problem constants (match reference) ----------------
#define BB    8
#define NP    2048
#define KNB   64
#define M1S   1024
#define M2S   256
#define R1SQ  0.04f   // 0.2^2
#define R2SQ  0.16f   // 0.4^2
#define BNEPS 1e-5f

typedef _Float16 half_t;
typedef __attribute__((ext_vector_type(16))) _Float16 v16h;
typedef __attribute__((ext_vector_type(8)))  float    v8f;

static inline int ceilDiv(long long a, long long b) { return (int)((a + b - 1) / b); }

// ---------------- WMMA GEMM, 16x16 tile per wave (fallback for small N) ------
__global__ void __launch_bounds__(256)
gemm_wmma_f16(const half_t* __restrict__ A, const half_t* __restrict__ WtT,
              const float* __restrict__ bias, float* __restrict__ C,
              int M, int N, int Kp, int Nbias) {
  const int lane = threadIdx.x & 31;
  const int wave = threadIdx.x >> 5;
  const int tile = blockIdx.x * 8 + wave;
  const int tilesN = N >> 4;
  const int totalTiles = (M >> 4) * tilesN;
  if (tile >= totalTiles) return;
  const int tm = tile / tilesN;
  const int tn = tile - tm * tilesN;

  const int rowA = (tm << 4) + (lane & 15);
  const int rowB = (tn << 4) + (lane & 15);
  const int kb   = (lane & 16) ? 8 : 0;   // lane-half K base per ISA A-fragment layout

  const unsigned int* __restrict__ Arow = (const unsigned int*)(A   + (size_t)rowA * Kp);
  const unsigned int* __restrict__ Brow = (const unsigned int*)(WtT + (size_t)rowB * Kp);

  v8f acc = {};
  for (int kk = 0; kk < Kp; kk += 32) {
    union { unsigned int u[8]; v16h v; } a, b;
    const int base = (kk + kb) >> 1;      // dword index of first half
#pragma unroll
    for (int j = 0; j < 4; ++j) {
      a.u[j]     = Arow[base + j];
      b.u[j]     = Brow[base + j];
      a.u[j + 4] = Arow[base + 8 + j];
      b.u[j + 4] = Brow[base + 8 + j];
    }
    acc = __builtin_amdgcn_wmma_f32_16x16x32_f16(
        false, a.v, false, b.v, (short)0, acc, false, false);
  }

  const int colC  = (tn << 4) + (lane & 15);
  const float bv  = (bias && colC < Nbias) ? bias[colC] : 0.f;
  const int rbase = (tm << 4) + ((lane & 16) ? 8 : 0);
#pragma unroll
  for (int r = 0; r < 8; ++r)
    C[(size_t)(rbase + r) * N + colC] = acc[r] + bv;
}

// ---------------- WMMA GEMM, 16x64 supertile per wave (N % 64 == 0) ----------
// A fragment loaded once per K-step, reused across 4 WMMAs (4x less A traffic).
__global__ void __launch_bounds__(256)
gemm_wmma_f16_n4(const half_t* __restrict__ A, const half_t* __restrict__ WtT,
                 const float* __restrict__ bias, float* __restrict__ C,
                 int M, int N, int Kp) {
  const int lane = threadIdx.x & 31;
  const int wave = threadIdx.x >> 5;
  const int tile = blockIdx.x * 8 + wave;
  const int tilesN = N >> 6;                 // 64-wide supertiles
  const int totalTiles = (M >> 4) * tilesN;
  if (tile >= totalTiles) return;
  const int tm  = tile / tilesN;
  const int tn4 = tile - tm * tilesN;

  const int rowA = (tm << 4) + (lane & 15);
  const int rowB = (tn4 << 6) + (lane & 15);
  const int kb   = (lane & 16) ? 8 : 0;

  const unsigned int* __restrict__ Arow = (const unsigned int*)(A   + (size_t)rowA * Kp);
  const unsigned int* __restrict__ Brow = (const unsigned int*)(WtT + (size_t)rowB * Kp);
  const size_t bs = (size_t)16 * Kp / 2;     // dword stride between B sub-tiles

  v8f acc0 = {}, acc1 = {}, acc2 = {}, acc3 = {};
  for (int kk = 0; kk < Kp; kk += 32) {
    const int base = (kk + kb) >> 1;
    union { unsigned int u[8]; v16h v; } a, b0, b1, b2, b3;
#pragma unroll
    for (int j = 0; j < 4; ++j) {
      a.u[j]      = Arow[base + j];
      a.u[j + 4]  = Arow[base + 8 + j];
      b0.u[j]     = Brow[base + j];
      b0.u[j + 4] = Brow[base + 8 + j];
      b1.u[j]     = Brow[bs + base + j];
      b1.u[j + 4] = Brow[bs + base + 8 + j];
      b2.u[j]     = Brow[2 * bs + base + j];
      b2.u[j + 4] = Brow[2 * bs + base + 8 + j];
      b3.u[j]     = Brow[3 * bs + base + j];
      b3.u[j + 4] = Brow[3 * bs + base + 8 + j];
    }
    if (kk + 32 < Kp) {                      // prefetch next K tile of A
      __builtin_prefetch(Arow + base + 16, 0, 3);
    }
    acc0 = __builtin_amdgcn_wmma_f32_16x16x32_f16(
        false, a.v, false, b0.v, (short)0, acc0, false, false);
    acc1 = __builtin_amdgcn_wmma_f32_16x16x32_f16(
        false, a.v, false, b1.v, (short)0, acc1, false, false);
    acc2 = __builtin_amdgcn_wmma_f32_16x16x32_f16(
        false, a.v, false, b2.v, (short)0, acc2, false, false);
    acc3 = __builtin_amdgcn_wmma_f32_16x16x32_f16(
        false, a.v, false, b3.v, (short)0, acc3, false, false);
  }

  const int rbase = (tm << 4) + ((lane & 16) ? 8 : 0);
  const int col0  = (tn4 << 6) + (lane & 15);
  v8f accs[4] = {acc0, acc1, acc2, acc3};
#pragma unroll
  for (int t = 0; t < 4; ++t) {
    const int colC = col0 + (t << 4);
    const float bv = bias ? bias[colC] : 0.f;
#pragma unroll
    for (int r = 0; r < 8; ++r)
      C[(size_t)(rbase + r) * N + colC] = accs[t][r] + bv;
  }
}

// ---------------- weight convert: f32 [Kreal,Nreal] -> f16 transposed [Np,Kp] --
__global__ void conv_weight(const float* __restrict__ W, half_t* __restrict__ out,
                            int Kreal, int Nreal, int Kp, int Np) {
  int i = blockIdx.x * 256 + threadIdx.x;
  if (i >= Np * Kp) return;
  int n = i / Kp, k = i - n * Kp;
  float v = (k < Kreal && n < Nreal) ? W[(size_t)k * Nreal + n] : 0.f;
  out[i] = (half_t)v;
}

// ---------------- farthest point sampling: 1 block per cloud -----------------
__global__ void __launch_bounds__(256)
fps_kernel(const float* __restrict__ pts, int n, int m, int* __restrict__ idxOut) {
  __shared__ float dist[NP];
  __shared__ float redV[256];
  __shared__ int   redI[256];
  __shared__ int   sLast;
  const int b = blockIdx.x;
  const int t = threadIdx.x;
  const float* P = pts + (size_t)b * n * 3;
  for (int i = t; i < n; i += 256) dist[i] = 3.4e38f;
  if (t == 0) sLast = 0;
  __syncthreads();
  for (int j = 0; j < m; ++j) {
    const int last = sLast;
    if (t == 0) idxOut[b * m + j] = last;       // emit carry (like lax.scan)
    const float lx = P[last * 3 + 0], ly = P[last * 3 + 1], lz = P[last * 3 + 2];
    float bestV = -1.f; int bestI = 0;
    for (int i = t; i < n; i += 256) {
      float dx = P[i * 3 + 0] - lx, dy = P[i * 3 + 1] - ly, dz = P[i * 3 + 2] - lz;
      float dd = fminf(dist[i], dx * dx + dy * dy + dz * dz);
      dist[i] = dd;
      if (dd > bestV) { bestV = dd; bestI = i; }
    }
    redV[t] = bestV; redI[t] = bestI;
    __syncthreads();
    for (int s = 128; s > 0; s >>= 1) {
      if (t < s) {
        if (redV[t + s] > redV[t] ||
            (redV[t + s] == redV[t] && redI[t + s] < redI[t])) {
          redV[t] = redV[t + s]; redI[t] = redI[t + s];
        }
      }
      __syncthreads();
    }
    if (t == 0) sLast = redI[0];
    __syncthreads();
  }
}

// ---------------- gather sampled centers -------------------------------------
__global__ void gather_centers(const float* __restrict__ pts, const int* __restrict__ idx,
                               float* __restrict__ centers, int n, int m) {
  int i = blockIdx.x * 256 + threadIdx.x;
  if (i >= BB * m) return;
  int b = i / m;
  int s = idx[i];
  const float* p = pts + ((size_t)b * n + s) * 3;
  centers[i * 3 + 0] = p[0];
  centers[i * 3 + 1] = p[1];
  centers[i * 3 + 2] = p[2];
}

// ---------------- radius query: first K in-ball by ascending index -----------
__global__ void radius_kernel(const float* __restrict__ pts, const float* __restrict__ centers,
                              int n, int m, float r2, int* __restrict__ nidx,
                              int* __restrict__ mask) {
  int i = blockIdx.x * 256 + threadIdx.x;   // center id over BB*m
  if (i >= BB * m) return;
  int b = i / m;
  const float* P = pts + (size_t)b * n * 3;
  const float cx = centers[i * 3 + 0], cy = centers[i * 3 + 1], cz = centers[i * 3 + 2];
  int cnt = 0;
  for (int p = 0; p < n && cnt < KNB; ++p) {
    float dx = P[p * 3 + 0] - cx, dy = P[p * 3 + 1] - cy, dz = P[p * 3 + 2] - cz;
    if (dx * dx + dy * dy + dz * dz <= r2) {
      nidx[(size_t)i * KNB + cnt] = p;
      ++cnt;
    }
  }
  for (int j = cnt; j < KNB; ++j) nidx[(size_t)i * KNB + j] = 0;
  for (int j = 0; j < KNB; ++j) mask[(size_t)i * KNB + j] = (j < cnt) ? 1 : 0;
}

// ---------------- edge feature builders --------------------------------------
__global__ void build_feat1(const float* __restrict__ pts, const float* __restrict__ centers,
                            const int* __restrict__ nidx, half_t* __restrict__ act,
                            int n, int m) {
  int r = blockIdx.x * 256 + threadIdx.x;           // edge row, rows = BB*m*KNB
  if (r >= BB * m * KNB) return;
  int c = r / KNB;
  int b = c / m;
  int p = nidx[r];
  const float* P = pts + ((size_t)b * n + p) * 3;
  half_t* o = act + (size_t)r * 32;
  o[0] = (half_t)(P[0] - centers[c * 3 + 0]);
  o[1] = (half_t)(P[1] - centers[c * 3 + 1]);
  o[2] = (half_t)(P[2] - centers[c * 3 + 2]);
  for (int j = 3; j < 32; ++j) o[j] = (half_t)0.f;
}

__global__ void build_feat2(const float* __restrict__ p1, const float* __restrict__ centers2,
                            const int* __restrict__ nidx, const half_t* __restrict__ x1h,
                            half_t* __restrict__ act) {
  int r = blockIdx.x * 256 + threadIdx.x;           // rows = BB*M2S*KNB
  if (r >= BB * M2S * KNB) return;
  int c = r / KNB;
  int b = c / M2S;
  int p = nidx[r];
  const half_t* f = x1h + ((size_t)b * M1S + p) * 128;
  half_t* o = act + (size_t)r * 160;
  for (int j = 0; j < 128; ++j) o[j] = f[j];
  const float* P = p1 + ((size_t)b * M1S + p) * 3;
  o[128] = (half_t)(P[0] - centers2[c * 3 + 0]);
  o[129] = (half_t)(P[1] - centers2[c * 3 + 1]);
  o[130] = (half_t)(P[2] - centers2[c * 3 + 2]);
  for (int j = 131; j < 160; ++j) o[j] = (half_t)0.f;
}

__global__ void build_feat3(const float* __restrict__ x2, const float* __restrict__ p2,
                            half_t* __restrict__ act) {
  int r = blockIdx.x * 256 + threadIdx.x;           // rows = BB*M2S
  if (r >= BB * M2S) return;
  half_t* o = act + (size_t)r * 288;
  for (int j = 0; j < 256; ++j) o[j] = (half_t)x2[(size_t)r * 256 + j];
  o[256] = (half_t)p2[r * 3 + 0];
  o[257] = (half_t)p2[r * 3 + 1];
  o[258] = (half_t)p2[r * 3 + 2];
  for (int j = 259; j < 288; ++j) o[j] = (half_t)0.f;
}

// ---------------- batch norm (training-mode, masked) -------------------------
__global__ void zero_f32(float* __restrict__ p, int n) {
  int i = blockIdx.x * 256 + threadIdx.x;
  if (i < n) p[i] = 0.f;
}

__global__ void __launch_bounds__(256)
bn_accum(const float* __restrict__ h, const int* __restrict__ mask,
         float* __restrict__ sum, float* __restrict__ sumsq, float* __restrict__ cnt,
         int rows, int C) {
  const int c = blockIdx.x;
  const int t = threadIdx.x;
  __shared__ float s1[256], s2[256], s3[256];
  float a1 = 0.f, a2 = 0.f, a3 = 0.f;
  for (int r = blockIdx.y * 256 + t; r < rows; r += gridDim.y * 256) {
    float w = mask ? (float)mask[r] : 1.f;
    float v = h[(size_t)r * C + c];
    a1 += v * w; a2 += v * v * w; a3 += w;
  }
  s1[t] = a1; s2[t] = a2; s3[t] = a3;
  __syncthreads();
  for (int s = 128; s > 0; s >>= 1) {
    if (t < s) { s1[t] += s1[t + s]; s2[t] += s2[t + s]; s3[t] += s3[t + s]; }
    __syncthreads();
  }
  if (t == 0) {
    atomicAdd(&sum[c], s1[0]);
    atomicAdd(&sumsq[c], s2[0]);
    if (c == 0) atomicAdd(cnt, s3[0]);
  }
}

__global__ void bn_relu_apply(const float* __restrict__ h, const int* __restrict__ mask,
                              const float* __restrict__ sum, const float* __restrict__ sumsq,
                              const float* __restrict__ cnt, half_t* __restrict__ act,
                              int rows, int C, int Kp) {
  long long i = (long long)blockIdx.x * 256 + threadIdx.x;   // over rows*Kp
  if (i >= (long long)rows * Kp) return;
  int r = (int)(i / Kp), c = (int)(i - (long long)r * Kp);
  float y = 0.f;
  if (c < C) {
    float n   = cnt[0];
    float mu  = sum[c] / n;
    float var = sumsq[c] / n - mu * mu;
    float v   = (h[(size_t)r * C + c] - mu) * rsqrtf(var + BNEPS);
    y = fmaxf(v, 0.f);
    if (mask && !mask[r]) y = 0.f;    // dead edge slots: excluded downstream anyway
  }
  act[i] = (half_t)y;
}

__global__ void relu_f16_apply(const float* __restrict__ h, half_t* __restrict__ act,
                               int rows, int C, int Kp) {
  long long i = (long long)blockIdx.x * 256 + threadIdx.x;
  if (i >= (long long)rows * Kp) return;
  int r = (int)(i / Kp), c = (int)(i - (long long)r * Kp);
  float y = (c < C) ? fmaxf(h[(size_t)r * C + c], 0.f) : 0.f;
  act[i] = (half_t)y;
}

// ---------------- masked max aggregation over K neighbors --------------------
__global__ void max_aggr(const float* __restrict__ h, const int* __restrict__ mask,
                         float* __restrict__ x, half_t* __restrict__ xh,
                         int centers, int C) {
  int i = blockIdx.x * 256 + threadIdx.x;   // over centers*C
  if (i >= centers * C) return;
  int cc = i / C, c = i - cc * C;
  float best = -3.4e38f;
  for (int k = 0; k < KNB; ++k)
    if (mask[(size_t)cc * KNB + k])
      best = fmaxf(best, h[((size_t)cc * KNB + k) * C + c]);
  x[i] = best;
  if (xh) xh[i] = (half_t)best;
}

__global__ void global_max(const float* __restrict__ h, float* __restrict__ g) {
  int i = blockIdx.x * 256 + threadIdx.x;   // BB*1024
  if (i >= BB * 1024) return;
  int b = i / 1024, c = i - b * 1024;
  float best = -3.4e38f;
  for (int m = 0; m < M2S; ++m)
    best = fmaxf(best, h[((size_t)b * M2S + m) * 1024 + c]);
  g[i] = best;
}

__global__ void build_headin(const float* __restrict__ g, half_t* __restrict__ act) {
  int i = blockIdx.x * 256 + threadIdx.x;   // 16*1024
  if (i >= 16 * 1024) return;
  int r = i / 1024, c = i - r * 1024;
  act[i] = (half_t)((r < BB) ? g[r * 1024 + c] : 0.f);
}

__global__ void logsoftmax_kernel(const float* __restrict__ h, float* __restrict__ out) {
  int b = threadIdx.x;
  if (b >= BB) return;
  const float* row = h + (size_t)b * 16;
  float mx = row[0];
  for (int j = 1; j < 10; ++j) mx = fmaxf(mx, row[j]);
  float s = 0.f;
  for (int j = 0; j < 10; ++j) s += expf(row[j] - mx);
  float ls = logf(s);
  for (int j = 0; j < 10; ++j) out[b * 10 + j] = row[j] - mx - ls;
}

// =============================================================================
extern "C" void kernel_launch(void* const* d_in, const int* in_sizes, int n_in,
                              void* d_out, int out_size, void* d_ws, size_t ws_size,
                              hipStream_t stream) {
  (void)in_sizes; (void)n_in; (void)out_size; (void)ws_size;

  const float* pos = (const float*)d_in[0];
  // params flatten in setup_inputs dict order: sa1(3 layers) sa2 sa3 head, W then b
  struct L { const float* W; const float* b; int Kr, Nr, Kp, Np; };
  L ly[12] = {
    {(const float*)d_in[1],  (const float*)d_in[2],     3,   64,   32,   64},
    {(const float*)d_in[3],  (const float*)d_in[4],    64,   64,   64,   64},
    {(const float*)d_in[5],  (const float*)d_in[6],    64,  128,   64,  128},
    {(const float*)d_in[7],  (const float*)d_in[8],   131,  128,  160,  128},
    {(const float*)d_in[9],  (const float*)d_in[10],  128,  128,  128,  128},
    {(const float*)d_in[11], (const float*)d_in[12],  128,  256,  128,  256},
    {(const float*)d_in[13], (const float*)d_in[14],  259,  256,  288,  256},
    {(const float*)d_in[15], (const float*)d_in[16],  256,  512,  256,  512},
    {(const float*)d_in[17], (const float*)d_in[18],  512, 1024,  512, 1024},
    {(const float*)d_in[19], (const float*)d_in[20], 1024,  512, 1024,  512},
    {(const float*)d_in[21], (const float*)d_in[22],  512,  256,  512,  256},
    {(const float*)d_in[23], (const float*)d_in[24],  256,   10,  256,   16},
  };
  float* out = (float*)d_out;

  const int rows1 = BB * M1S * KNB;   // 524288
  const int rows2 = BB * M2S * KNB;   // 131072
  const int rows3 = BB * M2S;         // 2048

  // ---- workspace bump allocator ----
  char* base = (char*)d_ws;
  size_t off = 0;
  auto alloc = [&](size_t bytes) -> void* {
    off = (off + 255) & ~(size_t)255;
    void* p = base + off;
    off += bytes;
    return p;
  };
  int*    idx1     = (int*)   alloc((size_t)BB * M1S * 4);
  float*  centers1 = (float*) alloc((size_t)BB * M1S * 3 * 4);
  int*    nidx1    = (int*)   alloc((size_t)rows1 * 4);
  int*    mask1    = (int*)   alloc((size_t)rows1 * 4);
  int*    idx2     = (int*)   alloc((size_t)BB * M2S * 4);
  float*  centers2 = (float*) alloc((size_t)BB * M2S * 3 * 4);
  int*    nidx2    = (int*)   alloc((size_t)rows2 * 4);
  int*    mask2    = (int*)   alloc((size_t)rows2 * 4);
  half_t* act      = (half_t*)alloc((size_t)rows1 * 64 * 2);     // max act footprint
  float*  h        = (float*) alloc((size_t)rows1 * 128 * 4);    // max h footprint
  float*  x1       = (float*) alloc((size_t)BB * M1S * 128 * 4);
  half_t* x1h      = (half_t*)alloc((size_t)BB * M1S * 128 * 2);
  float*  x2       = (float*) alloc((size_t)BB * M2S * 256 * 4);
  float*  g        = (float*) alloc((size_t)BB * 1024 * 4);
  float*  bnsum    = (float*) alloc(1024 * 4);
  float*  bnsq     = (float*) alloc(1024 * 4);
  float*  bncnt    = (float*) alloc(4);
  half_t* headin   = (half_t*)alloc((size_t)16 * 1024 * 2);
  half_t* wt[12];
  for (int i = 0; i < 12; ++i) wt[i] = (half_t*)alloc((size_t)ly[i].Np * ly[i].Kp * 2);

  // ---- convert weights to padded transposed f16 ----
  for (int i = 0; i < 12; ++i) {
    int n = ly[i].Np * ly[i].Kp;
    conv_weight<<<ceilDiv(n, 256), 256, 0, stream>>>(ly[i].W, wt[i],
                                                     ly[i].Kr, ly[i].Nr, ly[i].Kp, ly[i].Np);
  }

  auto gemm = [&](const half_t* A, int li, float* Cout, int M) {
    const int Np = ly[li].Np;
    if ((Np & 63) == 0) {
      int tiles = (M >> 4) * (Np >> 6);
      gemm_wmma_f16_n4<<<ceilDiv(tiles, 8), 256, 0, stream>>>(
          A, wt[li], ly[li].b, Cout, M, Np, ly[li].Kp);
    } else {
      int tiles = (M >> 4) * (Np >> 4);
      gemm_wmma_f16<<<ceilDiv(tiles, 8), 256, 0, stream>>>(
          A, wt[li], ly[li].b, Cout, M, Np, ly[li].Kp, ly[li].Nr);
    }
  };
  auto bn = [&](const float* hin, const int* mask, half_t* aout, int rows, int C, int Kp) {
    zero_f32<<<ceilDiv(C, 256), 256, 0, stream>>>(bnsum, C);
    zero_f32<<<ceilDiv(C, 256), 256, 0, stream>>>(bnsq, C);
    zero_f32<<<1, 256, 0, stream>>>(bncnt, 1);
    int rb = ceilDiv(rows, 256); if (rb > 64) rb = 64;
    bn_accum<<<dim3(C, rb), 256, 0, stream>>>(hin, mask, bnsum, bnsq, bncnt, rows, C);
    long long n = (long long)rows * Kp;
    bn_relu_apply<<<ceilDiv(n, 256), 256, 0, stream>>>(hin, mask, bnsum, bnsq, bncnt,
                                                       aout, rows, C, Kp);
  };

  // ================= SA1 =================
  fps_kernel<<<BB, 256, 0, stream>>>(pos, NP, M1S, idx1);
  gather_centers<<<ceilDiv(BB * M1S, 256), 256, 0, stream>>>(pos, idx1, centers1, NP, M1S);
  radius_kernel<<<ceilDiv(BB * M1S, 256), 256, 0, stream>>>(pos, centers1, NP, M1S, R1SQ,
                                                            nidx1, mask1);
  build_feat1<<<ceilDiv(rows1, 256), 256, 0, stream>>>(pos, centers1, nidx1, act, NP, M1S);

  gemm(act, 0, h, rows1);                       // 3->64
  bn(h, mask1, act, rows1, 64, 64);
  gemm(act, 1, h, rows1);                       // 64->64
  bn(h, mask1, act, rows1, 64, 64);
  gemm(act, 2, h, rows1);                       // 64->128 (plain)
  max_aggr<<<ceilDiv(BB * M1S * 128, 256), 256, 0, stream>>>(h, mask1, x1, x1h,
                                                             BB * M1S, 128);

  // ================= SA2 =================
  fps_kernel<<<BB, 256, 0, stream>>>(centers1, M1S, M2S, idx2);
  gather_centers<<<ceilDiv(BB * M2S, 256), 256, 0, stream>>>(centers1, idx2, centers2,
                                                             M1S, M2S);
  radius_kernel<<<ceilDiv(BB * M2S, 256), 256, 0, stream>>>(centers1, centers2, M1S, M2S,
                                                            R2SQ, nidx2, mask2);
  build_feat2<<<ceilDiv(rows2, 256), 256, 0, stream>>>(centers1, centers2, nidx2, x1h, act);

  gemm(act, 3, h, rows2);                       // 131->128
  bn(h, mask2, act, rows2, 128, 128);
  gemm(act, 4, h, rows2);                       // 128->128
  bn(h, mask2, act, rows2, 128, 128);
  gemm(act, 5, h, rows2);                       // 128->256 (plain)
  max_aggr<<<ceilDiv(BB * M2S * 256, 256), 256, 0, stream>>>(h, mask2, x2, nullptr,
                                                             BB * M2S, 256);

  // ================= SA3 (global MLP) =================
  build_feat3<<<ceilDiv(rows3, 256), 256, 0, stream>>>(x2, centers2, act);
  gemm(act, 6, h, rows3);                       // 259->256
  bn(h, nullptr, act, rows3, 256, 256);
  gemm(act, 7, h, rows3);                       // 256->512
  bn(h, nullptr, act, rows3, 512, 512);
  gemm(act, 8, h, rows3);                       // 512->1024 (plain)
  global_max<<<ceilDiv(BB * 1024, 256), 256, 0, stream>>>(h, g);

  // ================= Head =================
  build_headin<<<ceilDiv(16 * 1024, 256), 256, 0, stream>>>(g, headin);
  gemm(headin, 9, h, 16);                       // 1024->512
  relu_f16_apply<<<ceilDiv(16 * 512, 256), 256, 0, stream>>>(h, headin, 16, 512, 512);
  gemm(headin, 10, h, 16);                      // 512->256
  relu_f16_apply<<<ceilDiv(16 * 256, 256), 256, 0, stream>>>(h, headin, 16, 256, 256);
  gemm(headin, 11, h, 16);                      // 256->10 (padded 16)
  logsoftmax_kernel<<<1, 32, 0, stream>>>(h, out);
}
